// MultiHeadAttentionWithRoPE_48438641164767
// MI455X (gfx1250) — compile-verified, matrix operand order checked by reading, not run on hardware
//
#include <hip/hip_runtime.h>
#include <hip/hip_bf16.h>

typedef __attribute__((ext_vector_type(16))) __bf16 v16bf;
typedef __attribute__((ext_vector_type(8)))  float  v8f;

constexpr int Bc = 2, Tc = 2048, Cc = 2048, Hc = 32, KVHc = 8, Dc = 64;
constexpr int Gc = Hc / KVHc;

union FragU { v16bf v; uint4 q[2]; };

// Load a 16x32 bf16 WMMA A/B fragment.
// ISA layout: lane l -> row (l&15); K groups {8*(l>>4)..+7} in v0-3, {+16..+23} in v4-7.
__device__ __forceinline__ v16bf load_frag(const __bf16* base, int ld, int row0, int k0) {
  const int lane = threadIdx.x & 31;
  const __bf16* p = base + (size_t)(row0 + (lane & 15)) * (size_t)ld
                         + (size_t)(k0 + ((lane >> 4) << 3));
  FragU f;
  f.q[0] = *(const uint4*)(p);
  f.q[1] = *(const uint4*)(p + 16);
  return f.v;
}

// ---------------- fp32 -> bf16 conversion ----------------
__global__ void f32_to_bf16_kernel(const float* __restrict__ in, __bf16* __restrict__ out, size_t n) {
  size_t i = (size_t)blockIdx.x * blockDim.x + threadIdx.x;
  size_t stride = (size_t)gridDim.x * blockDim.x;
  for (; i < n; i += stride) out[i] = (__bf16)in[i];
}

// ---------------- GEMM: out[M,N] = A[M,K] @ W[N,K]^T ----------------
// Block = 256 threads (8 waves) arranged 2(M) x 4(N); block tile 64(M) x 256(N).
// Each wave computes 32x64: 2 A-frags x 4 B-frags -> 8 independent WMMAs per K-step.
template<bool OUT_BF16>
__global__ __launch_bounds__(256) void gemm_xWt(const __bf16* __restrict__ A,
                                                const __bf16* __restrict__ W,
                                                void* __restrict__ outp,
                                                int M, int N, int K) {
  const int wave = threadIdx.x >> 5;
  const int lane = threadIdx.x & 31;
  const int row0 = blockIdx.y * 64 + (wave & 1) * 32;
  const int col0 = blockIdx.x * 256 + (wave >> 1) * 64;

  v8f acc[2][4] = {};
  for (int k0 = 0; k0 < K; k0 += 32) {
    v16bf a0 = load_frag(A, K, row0, k0);
    v16bf a1 = load_frag(A, K, row0 + 16, k0);
    if (k0 + 32 < K) {
      const __bf16* pn = A + (size_t)(row0 + (lane & 15)) * (size_t)K + (k0 + 32);
      __builtin_prefetch(pn, 0, 0);   // global_prefetch_b8
    }
#pragma unroll
    for (int j = 0; j < 4; ++j) {
      v16bf b = load_frag(W, K, col0 + 16 * j, k0);
      acc[0][j] = __builtin_amdgcn_wmma_f32_16x16x32_bf16(false, a0, false, b,
                                                          (short)0, acc[0][j], false, false);
      acc[1][j] = __builtin_amdgcn_wmma_f32_16x16x32_bf16(false, a1, false, b,
                                                          (short)0, acc[1][j], false, false);
    }
  }
  const int half8 = (lane >> 4) << 3;   // C layout: lanes 16-31 hold rows +8
  const int cl = lane & 15;
#pragma unroll
  for (int m = 0; m < 2; ++m) {
#pragma unroll
    for (int j = 0; j < 4; ++j) {
#pragma unroll
      for (int r = 0; r < 8; ++r) {
        size_t row = (size_t)(row0 + 16 * m + r + half8);
        size_t col = (size_t)(col0 + 16 * j + cl);
        if constexpr (OUT_BF16) ((__bf16*)outp)[row * N + col] = (__bf16)acc[m][j][r];
        else                    ((float*)outp)[row * N + col]  = acc[m][j][r];
      }
    }
  }
}

// ---------------- RoPE (in-place on bf16 Q or K) ----------------
// post_scale folds the attention softmax scale 1/sqrt(D) into Q.
__global__ void rope_kernel(__bf16* __restrict__ qk, const float* __restrict__ cosb,
                            const float* __restrict__ sinb, int heads, size_t total,
                            float post_scale) {
  size_t i = (size_t)blockIdx.x * blockDim.x + threadIdx.x;
  if (i >= total) return;
  int d = (int)(i & 31);
  int h = (int)((i >> 5) % heads);
  size_t row = i / ((size_t)heads * 32);
  int t = (int)(row % Tc);
  __bf16* p = qk + row * ((size_t)heads * Dc) + (size_t)h * Dc;
  float x1 = (float)p[d];
  float x2 = (float)p[d + 32];
  float c0 = cosb[t * Dc + d],      s0 = sinb[t * Dc + d];
  float c1 = cosb[t * Dc + d + 32], s1 = sinb[t * Dc + d + 32];
  p[d]      = (__bf16)((x1 * c0 - x2 * s0) * post_scale);   // first half:  x1*cos - x2*sin
  p[d + 32] = (__bf16)((x2 * c1 + x1 * s1) * post_scale);   // second half: x2*cos + x1*sin
}

// ---------------- Flash attention (causal, GQA) ----------------
// grid = (T/128, H, B); 8 waves, each owns 16 q-rows. 64-key tiles.
// K tile staged once per block via global_load_async_to_lds_b128 (ASYNCcnt),
// V tile transposed into LDS, P relayout through LDS.
// NOTE: softmax scale 1/sqrt(D) is pre-folded into Q by rope_kernel.
__global__ __launch_bounds__(256) void flash_attn_kernel(const __bf16* __restrict__ Qb,
                                                         const __bf16* __restrict__ Kb,
                                                         const __bf16* __restrict__ Vb,
                                                         __bf16* __restrict__ Ob) {
  __shared__ __align__(16) __bf16 lds_k[64 * Dc];      // K tile, row-major [k][d]
  __shared__ __align__(16) __bf16 lds_vt[Dc * 64];     // V^T tile: Vt[d][k]
  __shared__ __align__(16) __bf16 lds_p[8 * 16 * 64];  // per-wave P scratch (C->A relayout)

  const int wave = threadIdx.x >> 5;
  const int lane = threadIdx.x & 31;
  const int half = lane >> 4;
  const int cl   = lane & 15;
  const int b    = blockIdx.z;
  const int h    = blockIdx.y;
  const int kvh  = h / Gc;
  const int qr0  = blockIdx.x * 128 + wave * 16;

  const __bf16* qbase = Qb + (size_t)b * Tc * Cc + (size_t)h * Dc;
  const __bf16* kbase = Kb + (size_t)b * Tc * (KVHc * Dc) + (size_t)kvh * Dc;
  const __bf16* vbase = Vb + (size_t)b * Tc * (KVHc * Dc) + (size_t)kvh * Dc;

  v16bf qa0 = load_frag(qbase, Cc, qr0, 0);
  v16bf qa1 = load_frag(qbase, Cc, qr0, 32);

  v8f oacc[4] = {};
  float mrow[8], lrow[8];
#pragma unroll
  for (int r = 0; r < 8; ++r) { mrow[r] = -1e30f; lrow[r] = 0.0f; }

  const int nkt = 2 * blockIdx.x + 2;   // 64-key tiles covering this block's causal span
  for (int kt = 0; kt < nkt; ++kt) {
    const int k0g = kt * 64;

    // --- async stage of K tile (64x64 bf16 = 8KB) into LDS: 512 x 16B chunks ---
#pragma unroll
    for (int c = 0; c < 2; ++c) {
      int elem = ((int)threadIdx.x + 256 * c) * 8;     // 8 bf16 per 16B chunk
      int kk = elem >> 6, dd = elem & 63;
      unsigned long long gaddr =
          (unsigned long long)(uintptr_t)(kbase + (size_t)(k0g + kk) * (KVHc * Dc) + dd);
      unsigned int laddr = (unsigned int)(uintptr_t)(&lds_k[kk * 64 + dd]);
      asm volatile("global_load_async_to_lds_b128 %0, %1, off"
                   :: "v"(laddr), "v"(gaddr) : "memory");
    }
    // --- cooperative transposed stage of V tile into LDS (ds_store_b16) ---
#pragma unroll
    for (int i = 0; i < 16; ++i) {
      int idx = threadIdx.x * 16 + i;
      int kk = idx >> 6, dd = idx & 63;
      lds_vt[dd * 64 + kk] = vbase[(size_t)(k0g + kk) * (KVHc * Dc) + dd];
    }
    asm volatile("s_wait_asynccnt 0" ::: "memory");
    __syncthreads();

    if (k0g <= qr0 + 15) {            // wave-uniform causal skip (EXEC stays full)
      // S = Q K^T : 16 q-rows x 64 keys, 8 WMMAs; K frags from LDS
      v8f s[4] = {};
#pragma unroll
      for (int j = 0; j < 4; ++j) {
        v16bf kb0 = load_frag(lds_k, 64, 16 * j, 0);
        s[j] = __builtin_amdgcn_wmma_f32_16x16x32_bf16(false, qa0, false, kb0,
                                                       (short)0, s[j], false, false);
        v16bf kb1 = load_frag(lds_k, 64, 16 * j, 32);
        s[j] = __builtin_amdgcn_wmma_f32_16x16x32_bf16(false, qa1, false, kb1,
                                                       (short)0, s[j], false, false);
      }
      // causal mask + per-row max (scale already folded into Q)
      float nm[8];
#pragma unroll
      for (int r = 0; r < 8; ++r) nm[r] = mrow[r];
#pragma unroll
      for (int j = 0; j < 4; ++j) {
        int col = k0g + 16 * j + cl;
#pragma unroll
        for (int r = 0; r < 8; ++r) {
          int rowq = qr0 + r + 8 * half;
          float sv = (col > rowq) ? -1e30f : s[j][r];
          s[j][r] = sv;
          nm[r] = fmaxf(nm[r], sv);
        }
      }
#pragma unroll
      for (int r = 0; r < 8; ++r)
#pragma unroll
        for (int off = 1; off < 16; off <<= 1)
          nm[r] = fmaxf(nm[r], __shfl_xor(nm[r], off, 32));  // 16-lane reduce, halves separate

      // P = exp(S - m), row sums, stage P to LDS in A-fragment-friendly layout
      float rs[8];
#pragma unroll
      for (int r = 0; r < 8; ++r) rs[r] = 0.0f;
      __bf16* pw = lds_p + wave * (16 * 64);
#pragma unroll
      for (int j = 0; j < 4; ++j) {
#pragma unroll
        for (int r = 0; r < 8; ++r) {
          float pv = __expf(s[j][r] - nm[r]);
          rs[r] += pv;
          pw[(r + 8 * half) * 64 + 16 * j + cl] = (__bf16)pv;
        }
      }
#pragma unroll
      for (int r = 0; r < 8; ++r) {
#pragma unroll
        for (int off = 1; off < 16; off <<= 1)
          rs[r] += __shfl_xor(rs[r], off, 32);
        float a = __expf(mrow[r] - nm[r]);       // rescale factor for old accum
        lrow[r] = lrow[r] * a + rs[r];
        mrow[r] = nm[r];
#pragma unroll
        for (int dj = 0; dj < 4; ++dj) oacc[dj][r] *= a;
      }
      // O += P @ V : 8 WMMAs, P and V^T fragments from LDS (ds_load_b128)
#pragma unroll
      for (int ks = 0; ks < 2; ++ks) {
        v16bf pa = load_frag(pw, 64, 0, 32 * ks);
#pragma unroll
        for (int dj = 0; dj < 4; ++dj) {
          v16bf vb = load_frag(lds_vt, 64, 16 * dj, 32 * ks);
          oacc[dj] = __builtin_amdgcn_wmma_f32_16x16x32_bf16(false, pa, false, vb,
                                                             (short)0, oacc[dj], false, false);
        }
      }
    }
    __syncthreads();
  }

  // normalize and write O (bf16) at column h*64
#pragma unroll
  for (int r = 0; r < 8; ++r) {
    float inv = 1.0f / lrow[r];
    int rowq = qr0 + r + 8 * half;
    __bf16* op = Ob + ((size_t)b * Tc + rowq) * Cc + (size_t)h * Dc;
#pragma unroll
    for (int dj = 0; dj < 4; ++dj)
      op[16 * dj + cl] = (__bf16)(oacc[dj][r] * inv);
  }
}

// ---------------- host launcher ----------------
extern "C" void kernel_launch(void* const* d_in, const int* in_sizes, int n_in,
                              void* d_out, int out_size, void* d_ws, size_t ws_size,
                              hipStream_t stream) {
  (void)in_sizes; (void)n_in; (void)out_size; (void)ws_size;
  const float* x    = (const float*)d_in[0];
  const float* Wq   = (const float*)d_in[1];
  const float* Wk   = (const float*)d_in[2];
  const float* Wv   = (const float*)d_in[3];
  const float* Wo   = (const float*)d_in[4];
  const float* cosb = (const float*)d_in[5];
  const float* sinb = (const float*)d_in[6];
  float* out = (float*)d_out;

  const size_t nX  = (size_t)Bc * Tc * Cc;         // 8388608
  const size_t nWq = (size_t)Cc * Cc;              // 4194304
  const size_t nWk = (size_t)KVHc * Dc * Cc;       // 1048576
  const size_t nKV = (size_t)Bc * Tc * KVHc * Dc;  // 2097152

  __bf16* ws  = (__bf16*)d_ws;
  __bf16* xb  = ws;
  __bf16* wqb = xb  + nX;
  __bf16* wkb = wqb + nWq;
  __bf16* wvb = wkb + nWk;
  __bf16* wob = wvb + nWk;
  __bf16* qb  = wob + nWq;
  __bf16* kb  = qb  + nX;
  __bf16* vb  = kb  + nKV;
  __bf16* ob  = vb  + nKV;

  auto cvt = [&](const float* src, __bf16* dst, size_t n) {
    int blocks = (int)((n + 255) / 256);
    if (blocks > 4096) blocks = 4096;
    f32_to_bf16_kernel<<<blocks, 256, 0, stream>>>(src, dst, n);
  };
  cvt(x, xb, nX); cvt(Wq, wqb, nWq); cvt(Wk, wkb, nWk); cvt(Wv, wvb, nWk); cvt(Wo, wob, nWq);

  dim3 blk(256);
  // Q = x Wq^T (4096x2048), K = x Wk^T, V = x Wv^T (4096x512)
  gemm_xWt<true ><<<dim3(2048 / 256, 4096 / 64), blk, 0, stream>>>(xb, wqb, qb, 4096, 2048, 2048);
  gemm_xWt<true ><<<dim3( 512 / 256, 4096 / 64), blk, 0, stream>>>(xb, wkb, kb, 4096,  512, 2048);
  gemm_xWt<true ><<<dim3( 512 / 256, 4096 / 64), blk, 0, stream>>>(xb, wvb, vb, 4096,  512, 2048);

  {
    // softmax scale 1/sqrt(D)=0.125 folded into Q here
    size_t nq = (size_t)Bc * Tc * Hc * (Dc / 2);
    rope_kernel<<<(int)((nq + 255) / 256), 256, 0, stream>>>(qb, cosb, sinb, Hc, nq, 0.125f);
    size_t nk = (size_t)Bc * Tc * KVHc * (Dc / 2);
    rope_kernel<<<(int)((nk + 255) / 256), 256, 0, stream>>>(kb, cosb, sinb, KVHc, nk, 1.0f);
  }

  flash_attn_kernel<<<dim3(Tc / 128, Hc, Bc), blk, 0, stream>>>(qb, kb, vb, ob);

  // out = O Wo^T (fp32 output)
  gemm_xWt<false><<<dim3(2048 / 256, 4096 / 64), blk, 0, stream>>>(ob, wob, out, 4096, 2048, 2048);
}
